// HybridBlock_14233521619272
// MI455X (gfx1250) — compile-verified, matrix operand order checked by reading, not run on hardware
//
#include <hip/hip_runtime.h>
#include <hip/hip_bf16.h>
#include <math.h>

// ---------------- types / constants ----------------
typedef __bf16 bf16_t;
typedef bf16_t v16bf __attribute__((ext_vector_type(16)));
typedef float  v8f   __attribute__((ext_vector_type(8)));
typedef unsigned int u32x4 __attribute__((ext_vector_type(4)));
typedef int i32x4 __attribute__((ext_vector_type(4)));
typedef int i32x8 __attribute__((ext_vector_type(8)));

#define D_      1024
#define LQ_     1024
#define LK_     2048
#define DINNER_ 2048
#define DSTATE_ 128
#define HM_     32
#define PDIM_   64
#define CONVD_  2304   // D_INNER + 2*D_STATE
#define DPROJ_  4384   // 2*D_INNER + 2*D_STATE + H_M
#define NH_     8
#define DH_     128
#define EPS_    1e-5f

union AV { v16bf v; uint4 u[2]; };

// Load 16 bf16 into the WMMA 16-bit operand layout:
// j=0..7  -> K = half8 + 0..7        (first b128)
// j=8..15 -> K = 16 + half8 + 0..7   (second b128)
__device__ __forceinline__ AV load16(const bf16_t* p, int half8) {
  AV r;
  r.u[0] = *reinterpret_cast<const uint4*>(p + half8);
  r.u[1] = *reinterpret_cast<const uint4*>(p + 16 + half8);
  return r;
}

__device__ __forceinline__ v8f wmma_bf(const AV& a, const AV& b, v8f c) {
  return __builtin_amdgcn_wmma_f32_16x16x32_bf16(
      false, a.v, false, b.v, (short)0, c, false, false);
}

// ---- CDNA5 async global->LDS copy (ASYNCcnt) ----
__device__ __forceinline__ void async_load_b128(unsigned int lds_addr, const void* gaddr) {
  asm volatile("global_load_async_to_lds_b128 %0, %1, off"
               :: "v"(lds_addr), "v"(gaddr) : "memory");
}
__device__ __forceinline__ void wait_asynccnt0() {
  asm volatile("s_wait_asynccnt 0x0" ::: "memory");
}

// ---- CDNA5 Tensor Data Mover: 2D tile global->LDS (TENSORcnt) ----
// data_size = 4B units. Tile: row_dwords x nrows, row stride = stride_dwords.
__device__ __forceinline__ void tdm_load_2d(unsigned int lds_addr, const void* gaddr,
                                            unsigned int row_dwords, unsigned int nrows,
                                            unsigned int stride_dwords) {
  unsigned long long ga = (unsigned long long)gaddr;
  u32x4 g0;
  g0[0] = 1u;                                   // count=1, user descriptor
  g0[1] = lds_addr;                             // lds_addr [63:32]
  g0[2] = (unsigned int)ga;                     // global_addr lo
  g0[3] = (unsigned int)((ga >> 32) & 0x01FFFFFFull) | (2u << 30);  // addr hi | type=2
  i32x8 g1;
  g1[0] = (int)(2u << 16);                      // data_size=4B; no multicast/pad/iterate
  g1[1] = (int)((stride_dwords & 0xFFFFu) << 16);          // tensor_dim0 lo16
  g1[2] = (int)((stride_dwords >> 16) | (nrows << 16));    // tensor_dim0 hi | tensor_dim1 lo
  g1[3] = (int)(row_dwords << 16);                         // tensor_dim1 hi | tile_dim0
  g1[4] = (int)nrows;                                      // tile_dim1 (tile_dim2=0)
  g1[5] = (int)stride_dwords;                              // tensor_dim0_stride lo32
  g1[6] = 0;                                               // stride hi | dim1_stride lo
  g1[7] = 0;
  i32x4 z4 = {0, 0, 0, 0};
  i32x8 z8 = {0, 0, 0, 0, 0, 0, 0, 0};
  // 6-arg form (clang-23 / therock headers): groups 0..3 + extra group + cpol
  __builtin_amdgcn_tensor_load_to_lds(g0, g1, z4, z4, z8, 0);
}

// Block-wide sum over 256 threads (8 waves of 32).
__device__ __forceinline__ float block_sum(float v, float* sbuf) {
  #pragma unroll
  for (int off = 16; off > 0; off >>= 1) v += __shfl_xor(v, off);
  int tid = threadIdx.x;
  if ((tid & 31) == 0) sbuf[tid >> 5] = v;
  __syncthreads();
  float r = (tid < 8) ? sbuf[tid] : 0.f;
  if (tid < 32) {
    #pragma unroll
    for (int off = 4; off > 0; off >>= 1) r += __shfl_xor(r, off);
  }
  if (tid == 0) sbuf[0] = r;
  __syncthreads();
  r = sbuf[0];
  __syncthreads();
  return r;
}

// ---------------- elementwise converts ----------------
__global__ __launch_bounds__(256) void cvt_bf16_kernel(
    const float* __restrict__ in, bf16_t* __restrict__ out, int n) {
  int i = blockIdx.x * 256 + threadIdx.x;
  if (i < n) out[i] = (bf16_t)in[i];
}

// in: K x N fp32 (row major), out: N x K bf16 (row major) -> WMMA B operand
__global__ __launch_bounds__(256) void tcvt_bf16_kernel(
    const float* __restrict__ in, bf16_t* __restrict__ out, int K, int N) {
  int i = blockIdx.x * 256 + threadIdx.x;
  if (i >= N * K) return;
  int n = i / K, k = i - n * K;
  out[i] = (bf16_t)in[(size_t)k * N + n];
}

// ---------------- norms ----------------
__global__ __launch_bounds__(256) void rmsnorm_bf16_kernel(
    const float* __restrict__ x, const float* __restrict__ w,
    bf16_t* __restrict__ out, int Dn) {
  __shared__ float sbuf[16];
  int row = blockIdx.x, tid = threadIdx.x;
  const float* xr = x + (size_t)row * Dn;
  int per = Dn / 256;
  float vals[8];
  float ss = 0.f;
  for (int i = 0; i < per; ++i) { float v = xr[tid + i * 256]; vals[i] = v; ss += v * v; }
  ss = block_sum(ss, sbuf);
  float inv = rsqrtf(ss / (float)Dn + EPS_);
  bf16_t* orow = out + (size_t)row * Dn;
  for (int i = 0; i < per; ++i) {
    int c = tid + i * 256;
    orow[c] = (bf16_t)(vals[i] * inv * w[c]);
  }
}

// u = rms(y * silu(z), gnorm_w), z = zxbcdt[:, :2048]
__global__ __launch_bounds__(256) void gated_rms_kernel(
    const float* __restrict__ y, const float* __restrict__ zx,
    const float* __restrict__ w, bf16_t* __restrict__ u) {
  __shared__ float sbuf[16];
  int row = blockIdx.x, tid = threadIdx.x;
  const float* yr = y + (size_t)row * DINNER_;
  const float* zr = zx + (size_t)row * DPROJ_;
  float g[8]; float ss = 0.f;
  #pragma unroll
  for (int i = 0; i < 8; ++i) {
    int c = tid + i * 256;
    float z = zr[c];
    float gv = yr[c] * (z / (1.f + __expf(-z)));
    g[i] = gv; ss += gv * gv;
  }
  ss = block_sum(ss, sbuf);
  float inv = rsqrtf(ss / (float)DINNER_ + EPS_);
  #pragma unroll
  for (int i = 0; i < 8; ++i) {
    int c = tid + i * 256;
    u[(size_t)row * DINNER_ + c] = (bf16_t)(g[i] * inv * w[c]);
  }
}

__global__ __launch_bounds__(256) void ln_bf16_kernel(
    const float* __restrict__ h, const float* __restrict__ w,
    const float* __restrict__ b, bf16_t* __restrict__ out) {
  __shared__ float sbuf[16];
  int row = blockIdx.x, tid = threadIdx.x;
  const float* hr = h + (size_t)row * D_;
  float v4[4]; float s = 0.f;
  #pragma unroll
  for (int i = 0; i < 4; ++i) { v4[i] = hr[tid + i * 256]; s += v4[i]; }
  s = block_sum(s, sbuf);
  float mu = s / (float)D_;
  float vs = 0.f;
  #pragma unroll
  for (int i = 0; i < 4; ++i) { float d = v4[i] - mu; vs += d * d; }
  vs = block_sum(vs, sbuf);
  float inv = rsqrtf(vs / (float)D_ + EPS_);
  #pragma unroll
  for (int i = 0; i < 4; ++i) {
    int c = tid + i * 256;
    out[(size_t)row * D_ + c] = (bf16_t)((v4[i] - mu) * inv * w[c] + b[c]);
  }
}

// ---------------- bf16 WMMA GEMM with TDM-staged B tiles ----------------
// C[M,N] = A[M,K] (bf16 row major) x Bt[N,K] (bf16 row major == B^T) (+ resid)
// 256 threads = 8 waves; wave tile 16x32; WG tile 128x32.
// B tile (32 rows x 32 k = 2KB) staged to LDS by the Tensor Data Mover,
// double buffered, synchronized via TENSORcnt + barrier.
__global__ __launch_bounds__(256) void gemm_bf16_kernel(
    const bf16_t* __restrict__ A, const bf16_t* __restrict__ Bt,
    const float* __restrict__ resid, float* __restrict__ C,
    int M, int N, int K) {
  __shared__ __align__(16) bf16_t Bsh[2][32 * 32];
  int wave = threadIdx.x >> 5, lane = threadIdx.x & 31;
  int lm = lane & 15, lh = lane >> 4, half8 = lh * 8;
  int mrow = blockIdx.y * 128 + wave * 16 + lm;
  int n0   = blockIdx.x * 32;
  const bf16_t* arow  = A  + (size_t)mrow * K;
  const bf16_t* btile = Bt + (size_t)n0 * K;      // tile row stride = K bf16
  unsigned int lds0 = (unsigned int)(size_t)&Bsh[0][0];
  unsigned int lds1 = (unsigned int)(size_t)&Bsh[1][0];

  v8f acc0, acc1;
  #pragma unroll
  for (int v = 0; v < 8; ++v) { acc0[v] = 0.f; acc1[v] = 0.f; }

  if (wave == 0)
    tdm_load_2d(lds0, btile, /*row_dwords=*/16, /*nrows=*/32, /*stride=*/(unsigned)(K >> 1));

  int buf = 0;
  for (int kb = 0; kb < K; kb += 32) {
    if (wave == 0) __builtin_amdgcn_s_wait_tensorcnt(0);
    __syncthreads();                       // staged tile visible to all waves
    if (wave == 0 && kb + 32 < K)
      tdm_load_2d(buf ? lds0 : lds1, btile + (kb + 32), 16, 32, (unsigned)(K >> 1));

    AV a = load16(arow + kb, half8);       // global_load_b128 x2
    const bf16_t* r0 = &Bsh[buf][lm * 32];
    const bf16_t* r1 = &Bsh[buf][(16 + lm) * 32];
    AV b0, b1;                             // ds_load_b128 x4
    b0.u[0] = *reinterpret_cast<const uint4*>(r0 + half8);
    b0.u[1] = *reinterpret_cast<const uint4*>(r0 + 16 + half8);
    b1.u[0] = *reinterpret_cast<const uint4*>(r1 + half8);
    b1.u[1] = *reinterpret_cast<const uint4*>(r1 + 16 + half8);
    acc0 = wmma_bf(a, b0, acc0);
    acc1 = wmma_bf(a, b1, acc1);
    buf ^= 1;
    __syncthreads();                       // reads done before next TDM overwrite
  }

  int orow = blockIdx.y * 128 + wave * 16 + lh * 8;
  #pragma unroll
  for (int v = 0; v < 8; ++v) {
    size_t i0 = (size_t)(orow + v) * N + n0 + lm;
    size_t i1 = i0 + 16;
    float r0 = resid ? resid[i0] : 0.f;
    float r1 = resid ? resid[i1] : 0.f;
    C[i0] = acc0[v] + r0;
    C[i1] = acc1[v] + r1;
  }
}

// ---------------- mamba pieces ----------------
__global__ __launch_bounds__(256) void conv_silu_kernel(
    const float* __restrict__ zx, const float* __restrict__ cw,
    const float* __restrict__ cb, float* __restrict__ xbc) {
  int i = blockIdx.x * 256 + threadIdx.x;
  if (i >= LQ_ * CONVD_) return;
  int l = i / CONVD_, c = i - l * CONVD_;
  float acc = cb[c];
  #pragma unroll
  for (int k = 0; k < 4; ++k) {
    int lp = l - 3 + k;
    if (lp >= 0) acc += zx[(size_t)lp * DPROJ_ + DINNER_ + c] * cw[c * 4 + k];
  }
  xbc[i] = acc / (1.f + __expf(-acc));   // silu
}

__global__ __launch_bounds__(256) void dt_decay_kernel(
    const float* __restrict__ zx, const float* __restrict__ dt_bias,
    const float* __restrict__ A_log, float* __restrict__ dt,
    float* __restrict__ dec) {
  int i = blockIdx.x * 256 + threadIdx.x;
  if (i >= LQ_ * HM_) return;
  int l = i / HM_, h = i - l * HM_;
  float raw = zx[(size_t)l * DPROJ_ + (DPROJ_ - HM_) + h] + dt_bias[h];
  float sp = (raw > 20.f) ? raw : log1pf(__expf(raw));
  float A = -__expf(A_log[h]);
  dt[i] = sp;
  dec[i] = __expf(sp * A);
}

// Sequential SSM scan; one WG per head. tid = p*4 + nq, each thread owns 32 n's.
__global__ __launch_bounds__(256) void ssm_scan_kernel(
    const float* __restrict__ xbc, const float* __restrict__ dt,
    const float* __restrict__ dec, const float* __restrict__ D_skip,
    float* __restrict__ y) {
  __shared__ float Bs[DSTATE_];
  __shared__ float Cs[DSTATE_];
  int h = blockIdx.x;
  int tid = threadIdx.x;
  int p = tid >> 2, nq = tid & 3, n0 = nq * 32;
  float hst[32];
  #pragma unroll
  for (int i = 0; i < 32; ++i) hst[i] = 0.f;
  float dsk = D_skip[h];
  for (int l = 0; l < LQ_; ++l) {
    __syncthreads();
    if (tid < 128)      Bs[tid]       = xbc[(size_t)l * CONVD_ + DINNER_ + tid];
    else                Cs[tid - 128] = xbc[(size_t)l * CONVD_ + DINNER_ + DSTATE_ + (tid - 128)];
    __syncthreads();
    float d  = dec[l * HM_ + h];
    float dv = dt[l * HM_ + h];
    float xv = xbc[(size_t)l * CONVD_ + h * PDIM_ + p];
    float dbx = dv * xv;
    float acc = 0.f;
    #pragma unroll
    for (int i = 0; i < 32; ++i) {
      float b = Bs[n0 + i], c = Cs[n0 + i];
      hst[i] = hst[i] * d + dbx * b;
      acc += hst[i] * c;
    }
    acc += __shfl_xor(acc, 1);
    acc += __shfl_xor(acc, 2);
    if (nq == 0) y[(size_t)l * DINNER_ + h * PDIM_ + p] = acc + dsk * xv;
  }
}

// ---------------- flash attention (WMMA, online softmax on S^T) ----------------
// grid (LQ/128, NH), block 256 (8 waves; each wave: 16 q rows, full dh=128).
// K tile staged to LDS with async global->LDS copies (shared by all 8 waves).
__global__ __launch_bounds__(256) void flash_attn_kernel(
    const bf16_t* __restrict__ qbf, const bf16_t* __restrict__ kbf,
    const bf16_t* __restrict__ vbf, const unsigned char* __restrict__ mask,
    float* __restrict__ ctx) {
  __shared__ __align__(16) bf16_t Ks[32 * DH_];  // row-major: Ks[k*128 + c]
  __shared__ __align__(16) bf16_t Vs[32 * DH_];  // col-major: Vs[c*32 + k]
  int head = blockIdx.y;
  int wave = threadIdx.x >> 5, lane = threadIdx.x & 31;
  int lm = lane & 15, lh = lane >> 4, half8 = lh * 8;
  int qrow0 = blockIdx.x * 128 + wave * 16;
  int hoff = head * DH_;
  const float sc = 0.08838834764831845f;  // 1/sqrt(128)
  unsigned int ksb = (unsigned int)(size_t)&Ks[0];

  // Q as B operand of S^T = K_tile x Q^T: lane holds q row qrow0+lm, 4 dh-blocks
  AV bq[4];
  const bf16_t* qrow = qbf + (size_t)(qrow0 + lm) * D_ + hoff;
  #pragma unroll
  for (int d = 0; d < 4; ++d) bq[d] = load16(qrow + d * 32, half8);

  v8f o[8];
  #pragma unroll
  for (int t = 0; t < 8; ++t)
    #pragma unroll
    for (int v = 0; v < 8; ++v) o[t][v] = 0.f;
  float rm = -3.0e38f, rl = 0.f;

  for (int kb = 0; kb < LK_; kb += 32) {
    __syncthreads();                       // prior iteration's LDS reads done
    if (wave == 0) {
      // async-stage K tile: 32 rows x 256B = 16 x (32 lanes x 16B)
      #pragma unroll
      for (int t = 0; t < 16; ++t) {
        int c = t * 32 + lane;
        int row = c >> 4, col = c & 15;
        async_load_b128(ksb + (unsigned)(row * 256 + col * 16),
                        kbf + (size_t)(kb + row) * D_ + hoff + col * 8);
      }
    }
    // stage V tile transposed (col-major) - all 8 waves
    for (int i = threadIdx.x; i < 32 * DH_; i += 256) {
      int k = i >> 7, c = i & 127;
      Vs[c * 32 + k] = vbf[(size_t)(kb + k) * D_ + hoff + c];
    }
    if (wave == 0) wait_asynccnt0();
    __syncthreads();                       // staged K/V visible

    // S^T two 16x16 tiles (kcols [0,16) and [16,32)): A = K rows (LDS), B = Q^T
    v8f s0, s1;
    #pragma unroll
    for (int v = 0; v < 8; ++v) { s0[v] = 0.f; s1[v] = 0.f; }
    #pragma unroll
    for (int d = 0; d < 4; ++d) {
      const bf16_t* kr0 = &Ks[lm * DH_ + d * 32];
      const bf16_t* kr1 = &Ks[(16 + lm) * DH_ + d * 32];
      AV a0, a1;
      a0.u[0] = *reinterpret_cast<const uint4*>(kr0 + half8);
      a0.u[1] = *reinterpret_cast<const uint4*>(kr0 + 16 + half8);
      a1.u[0] = *reinterpret_cast<const uint4*>(kr1 + half8);
      a1.u[1] = *reinterpret_cast<const uint4*>(kr1 + 16 + half8);
      s0 = wmma_bf(a0, bq[d], s0);
      s1 = wmma_bf(a1, bq[d], s1);
    }
    // scale + padding mask (S^T: element v is kcol = kb + v + 8*lh [+16])
    #pragma unroll
    for (int v = 0; v < 8; ++v) {
      int kc = kb + v + 8 * lh;
      s0[v] = s0[v] * sc + (mask[kc]      ? -1.0e9f : 0.f);
      s1[v] = s1[v] * sc + (mask[kc + 16] ? -1.0e9f : 0.f);
    }
    // online softmax; q index = lm (lanes L and L+16 pair via shfl_xor 16)
    float mx = s0[0];
    #pragma unroll
    for (int v = 0; v < 8; ++v) { mx = fmaxf(mx, s0[v]); mx = fmaxf(mx, s1[v]); }
    mx = fmaxf(mx, __shfl_xor(mx, 16));
    float mnew = fmaxf(rm, mx);
    float scale = __expf(rm - mnew);
    float lsum = 0.f;
    #pragma unroll
    for (int v = 0; v < 8; ++v) {
      s0[v] = __expf(s0[v] - mnew);
      s1[v] = __expf(s1[v] - mnew);
      lsum += s0[v] + s1[v];
    }
    lsum += __shfl_xor(lsum, 16);
    rl = rl * scale + lsum;
    rm = mnew;
    // rescale O: O row (q) for accum element v is v + 8*lh; stats live in lane q
    float sf[8];
    #pragma unroll
    for (int v = 0; v < 8; ++v) sf[v] = __shfl(scale, v + 8 * lh);
    #pragma unroll
    for (int t = 0; t < 8; ++t)
      #pragma unroll
      for (int v = 0; v < 8; ++v) o[t][v] *= sf[v];
    // P^T (D-layout) is already the A-operand layout for P x V: no shuffles
    AV pa;
    #pragma unroll
    for (int j = 0; j < 8; ++j) {
      pa.v[j]     = (bf16_t)s0[j];   // K = half8 + j       (kcols 0..15)
      pa.v[j + 8] = (bf16_t)s1[j];   // K = 16 + half8 + j  (kcols 16..31)
    }
    // O += P x V, B from LDS (col-major V -> contiguous K)
    #pragma unroll
    for (int t = 0; t < 8; ++t) {
      const bf16_t* col = &Vs[(t * 16 + lm) * 32];
      AV b;
      b.u[0] = *reinterpret_cast<const uint4*>(col + half8);
      b.u[1] = *reinterpret_cast<const uint4*>(col + 16 + half8);
      o[t] = wmma_bf(pa, b, o[t]);
    }
  }
  // normalize and write ctx (fp32)
  float inv = 1.f / rl;
  float rf[8];
  #pragma unroll
  for (int v = 0; v < 8; ++v) rf[v] = __shfl(inv, v + 8 * lh);
  #pragma unroll
  for (int t = 0; t < 8; ++t)
    #pragma unroll
    for (int v = 0; v < 8; ++v) {
      int q = qrow0 + v + 8 * lh;
      ctx[(size_t)q * D_ + hoff + t * 16 + lm] = o[t][v] * rf[v];
    }
}

// ---------------- host orchestration ----------------
extern "C" void kernel_launch(void* const* d_in, const int* in_sizes, int n_in,
                              void* d_out, int out_size, void* d_ws, size_t ws_size,
                              hipStream_t stream) {
  const float* x          = (const float*)d_in[0];
  const float* enc        = (const float*)d_in[1];
  const unsigned char* pm = (const unsigned char*)d_in[2];
  const float* m_norm_w   = (const float*)d_in[3];
  const float* in_proj_w  = (const float*)d_in[4];
  const float* conv_w     = (const float*)d_in[5];
  const float* conv_b     = (const float*)d_in[6];
  const float* dt_bias    = (const float*)d_in[7];
  const float* A_log      = (const float*)d_in[8];
  const float* D_skip     = (const float*)d_in[9];
  const float* gnorm_w    = (const float*)d_in[10];
  const float* out_proj_w = (const float*)d_in[11];
  const float* ca_ln_w    = (const float*)d_in[12];
  const float* ca_ln_b    = (const float*)d_in[13];
  const float* Wq         = (const float*)d_in[14];
  const float* Wk         = (const float*)d_in[15];
  const float* Wv         = (const float*)d_in[16];
  const float* Wo         = (const float*)d_in[17];
  float* out = (float*)d_out;

  char* base = (char*)d_ws; size_t off = 0;
  auto alloc = [&](size_t b) -> char* {
    char* p = base + off; off = (off + b + 255) & ~(size_t)255; return p;
  };
  bf16_t* xn1  = (bf16_t*)alloc((size_t)LQ_ * D_ * 2);
  bf16_t* w1t  = (bf16_t*)alloc((size_t)DPROJ_ * D_ * 2);
  float*  zx   = (float*) alloc((size_t)LQ_ * DPROJ_ * 4);
  float*  xbc  = (float*) alloc((size_t)LQ_ * CONVD_ * 4);
  float*  dtb  = (float*) alloc((size_t)LQ_ * HM_ * 4);
  float*  decb = (float*) alloc((size_t)LQ_ * HM_ * 4);
  float*  ybuf = (float*) alloc((size_t)LQ_ * DINNER_ * 4);
  bf16_t* ubf  = (bf16_t*)alloc((size_t)LQ_ * DINNER_ * 2);
  bf16_t* opt  = (bf16_t*)alloc((size_t)D_ * DINNER_ * 2);
  float*  hbuf = (float*) alloc((size_t)LQ_ * D_ * 4);
  bf16_t* xn2  = (bf16_t*)alloc((size_t)LQ_ * D_ * 2);
  bf16_t* wqt  = (bf16_t*)alloc((size_t)D_ * D_ * 2);
  bf16_t* wkt  = (bf16_t*)alloc((size_t)D_ * D_ * 2);
  bf16_t* wvt  = (bf16_t*)alloc((size_t)D_ * D_ * 2);
  bf16_t* wot  = (bf16_t*)alloc((size_t)D_ * D_ * 2);
  bf16_t* encb = (bf16_t*)alloc((size_t)LK_ * D_ * 2);
  float*  qf   = (float*) alloc((size_t)LQ_ * D_ * 4);
  float*  kf   = (float*) alloc((size_t)LK_ * D_ * 4);
  float*  vf   = (float*) alloc((size_t)LK_ * D_ * 4);
  bf16_t* qbf  = (bf16_t*)alloc((size_t)LQ_ * D_ * 2);
  bf16_t* kbf  = (bf16_t*)alloc((size_t)LK_ * D_ * 2);
  bf16_t* vbf  = (bf16_t*)alloc((size_t)LK_ * D_ * 2);
  float*  ctx  = (float*) alloc((size_t)LQ_ * D_ * 4);
  bf16_t* ctxb = (bf16_t*)alloc((size_t)LQ_ * D_ * 2);

  auto nb = [](size_t n) { return (unsigned)((n + 255) / 256); };

  // ---- Mamba2 block ----
  rmsnorm_bf16_kernel<<<LQ_, 256, 0, stream>>>(x, m_norm_w, xn1, D_);
  tcvt_bf16_kernel<<<nb((size_t)DPROJ_ * D_), 256, 0, stream>>>(in_proj_w, w1t, D_, DPROJ_);
  gemm_bf16_kernel<<<dim3(DPROJ_ / 32, LQ_ / 128), 256, 0, stream>>>(
      xn1, w1t, nullptr, zx, LQ_, DPROJ_, D_);
  conv_silu_kernel<<<nb((size_t)LQ_ * CONVD_), 256, 0, stream>>>(zx, conv_w, conv_b, xbc);
  dt_decay_kernel<<<nb((size_t)LQ_ * HM_), 256, 0, stream>>>(zx, dt_bias, A_log, dtb, decb);
  ssm_scan_kernel<<<HM_, 256, 0, stream>>>(xbc, dtb, decb, D_skip, ybuf);
  gated_rms_kernel<<<LQ_, 256, 0, stream>>>(ybuf, zx, gnorm_w, ubf);
  tcvt_bf16_kernel<<<nb((size_t)D_ * DINNER_), 256, 0, stream>>>(out_proj_w, opt, DINNER_, D_);
  gemm_bf16_kernel<<<dim3(D_ / 32, LQ_ / 128), 256, 0, stream>>>(
      ubf, opt, x, hbuf, LQ_, D_, DINNER_);

  // ---- Cross attention ----
  ln_bf16_kernel<<<LQ_, 256, 0, stream>>>(hbuf, ca_ln_w, ca_ln_b, xn2);
  tcvt_bf16_kernel<<<nb((size_t)D_ * D_), 256, 0, stream>>>(Wq, wqt, D_, D_);
  tcvt_bf16_kernel<<<nb((size_t)D_ * D_), 256, 0, stream>>>(Wk, wkt, D_, D_);
  tcvt_bf16_kernel<<<nb((size_t)D_ * D_), 256, 0, stream>>>(Wv, wvt, D_, D_);
  tcvt_bf16_kernel<<<nb((size_t)D_ * D_), 256, 0, stream>>>(Wo, wot, D_, D_);
  cvt_bf16_kernel<<<nb((size_t)LK_ * D_), 256, 0, stream>>>(enc, encb, LK_ * D_);

  gemm_bf16_kernel<<<dim3(D_ / 32, LQ_ / 128), 256, 0, stream>>>(
      xn2, wqt, nullptr, qf, LQ_, D_, D_);
  gemm_bf16_kernel<<<dim3(D_ / 32, LK_ / 128), 256, 0, stream>>>(
      encb, wkt, nullptr, kf, LK_, D_, D_);
  gemm_bf16_kernel<<<dim3(D_ / 32, LK_ / 128), 256, 0, stream>>>(
      encb, wvt, nullptr, vf, LK_, D_, D_);
  cvt_bf16_kernel<<<nb((size_t)LQ_ * D_), 256, 0, stream>>>(qf, qbf, LQ_ * D_);
  cvt_bf16_kernel<<<nb((size_t)LK_ * D_), 256, 0, stream>>>(kf, kbf, LK_ * D_);
  cvt_bf16_kernel<<<nb((size_t)LK_ * D_), 256, 0, stream>>>(vf, vbf, LK_ * D_);

  flash_attn_kernel<<<dim3(LQ_ / 128, NH_), 256, 0, stream>>>(qbf, kbf, vbf, pm, ctx);

  cvt_bf16_kernel<<<nb((size_t)LQ_ * D_), 256, 0, stream>>>(ctx, ctxb, LQ_ * D_);
  gemm_bf16_kernel<<<dim3(D_ / 32, LQ_ / 128), 256, 0, stream>>>(
      ctxb, wot, hbuf, out, LQ_, D_, D_);
}
